// GNNClassifierNoPos_88648124990643
// MI455X (gfx1250) — matrix operand order repaired and checked by reading, not compiled, not run on hardware
//
#include <hip/hip_runtime.h>
#include <hip/hip_bf16.h>

typedef __attribute__((ext_vector_type(16))) __bf16 v16bf;
typedef __attribute__((ext_vector_type(8)))  float  v8f;

#define HID 64

// x[i,f] = shape_W[sid[i],f] + col_W[cid[i],f]
__global__ void embed_kernel(const int* __restrict__ sid, const int* __restrict__ cid,
                             const float* __restrict__ sW, const float* __restrict__ cW,
                             float* __restrict__ x, int N) {
  int idx = blockIdx.x * blockDim.x + threadIdx.x;
  if (idx >= N * HID) return;
  int i = idx >> 6, f = idx & 63;
  x[idx] = sW[sid[i] * HID + f] + cW[cid[i] * HID + f];
}

// in-degree counts (float, for mean aggregation)
__global__ void count_kernel(const int* __restrict__ edges, float* __restrict__ cnt, int E) {
  int e = blockIdx.x * blockDim.x + threadIdx.x;
  if (e >= E) return;
  atomicAdd(&cnt[edges[E + e]], 1.0f);
}

// msg[dst,:] += x[src,:]  (one wave per edge, 2 floats per lane; 256B coalesced)
__global__ void scatter_kernel(const int* __restrict__ edges, const float* __restrict__ x,
                               float* __restrict__ msg, int E) {
  int idx = blockIdx.x * blockDim.x + threadIdx.x;
  if (idx >= E * 32) return;
  int e = idx >> 5;
  int f = (idx & 31) * 2;
  int src = edges[e], dst = edges[E + e];
  const float2 v = *(const float2*)(x + src * HID + f);
  atomicAdd(&msg[dst * HID + f + 0], v.x);
  atomicAdd(&msg[dst * HID + f + 1], v.y);
}

// Pack W1l,W1r,W2l,W2r (f32 row-major [k][n], 64x64) into bf16 WMMA-B fragment order.
// pb[(m*8 + ks*4 + t)*512 + lane*16 + i] = bf16(W[k][n]) with
//   k = ks*32 + (lane>=16 ? 16 : 0) + i,   n = t*16 + (lane&15)
__global__ void pack_w_kernel(const float* __restrict__ W1l, const float* __restrict__ W1r,
                              const float* __restrict__ W2l, const float* __restrict__ W2r,
                              __bf16* __restrict__ pb) {
  int p = blockIdx.x * blockDim.x + threadIdx.x;
  if (p >= 4 * 4096) return;
  int m      = p >> 12;          // 0:W1l 1:W1r 2:W2l 3:W2r
  int rem    = p & 4095;
  int fragid = rem >> 9;         // 0..7  (= ks*4 + t)
  int lane   = (rem >> 4) & 31;
  int i      = rem & 15;
  int ks = fragid >> 2, t = fragid & 3;
  int k = ks * 32 + ((lane & 16) ? 16 : 0) + i;
  int n = t * 16 + (lane & 15);
  const float* W = (m == 0) ? W1l : (m == 1) ? W1r : (m == 2) ? W2l : W2r;
  pb[p] = (__bf16)W[k * HID + n];
}

// out[16 rows x 64 cols] per wave:  mean @ Wl + x @ Wr + b   via v_wmma_f32_16x16x32_bf16.
// Packed bf16 weights (16KB: Wl frags then Wr frags) staged once per block into LDS.
__global__ void sage_wmma_kernel(const float* __restrict__ msg, const float* __restrict__ cnt,
                                 const float* __restrict__ x,
                                 const __bf16* __restrict__ pbw, const float* __restrict__ b,
                                 float* __restrict__ out, int N) {
  __shared__ alignas(32) __bf16 lw[8192];          // 16 KB of 320 KB/WGP
  for (int j = threadIdx.x; j < 1024; j += blockDim.x)
    ((uint4*)lw)[j] = ((const uint4*)pbw)[j];      // cooperative b128 copy
  __syncthreads();

  int wave = (blockIdx.x * blockDim.x + threadIdx.x) >> 5;
  int lane = threadIdx.x & 31;
  int ntiles = (N + 15) >> 4;
  if (wave >= ntiles) return;                      // after barrier; uniform per wave

  int row0 = wave << 4;
  int arow = row0 + (lane & 15);
  if (arow >= N) arow = N - 1;                     // clamp reads; stores are guarded
  float inv = 1.0f / fmaxf(cnt[arow], 1.0f);

  const int hiA = (lane >= 16) ? 16 : 0;           // A-frag K offset for upper lane half

  v8f zero = {0.f, 0.f, 0.f, 0.f, 0.f, 0.f, 0.f, 0.f};
  v8f acc[4] = {zero, zero, zero, zero};

#pragma unroll
  for (int ks = 0; ks < 2; ++ks) {                 // K = 64 -> two 32-wide WMMA K steps
    // A fragments: element i -> K = ks*32 + hiA + i
    v16bf amean, aself;
#pragma unroll
    for (int i = 0; i < 16; ++i) {
      int k = ks * 32 + hiA + i;
      amean[i] = (__bf16)(msg[arow * HID + k] * inv);
      aself[i] = (__bf16)(x[arow * HID + k]);
    }
#pragma unroll
    for (int t = 0; t < 4; ++t) {                  // four 16-wide N tiles
      int fragid = ks * 4 + t;
      const v16bf bl = *(const v16bf*)(lw + (fragid * 32 + lane) * 16);
      const v16bf br = *(const v16bf*)(lw + ((8 + fragid) * 32 + lane) * 16);
      acc[t] = __builtin_amdgcn_wmma_f32_16x16x32_bf16(false, amean, false, bl,
                                                       (short)0, acc[t], false, false);
      acc[t] = __builtin_amdgcn_wmma_f32_16x16x32_bf16(false, aself, false, br,
                                                       (short)0, acc[t], false, false);
    }
  }

#pragma unroll
  for (int t = 0; t < 4; ++t) {
    int n = t * 16 + (lane & 15);
    float bias = b[n];
#pragma unroll
    for (int v = 0; v < 8; ++v) {                  // C layout: M = v + 8*(lane>=16)
      int r = row0 + v + ((lane >> 4) << 3);
      if (r < N) out[r * HID + n] = acc[t][v] + bias;
    }
  }
}

// per-feature sum / sum-of-squares over node dim (LDS-reduced, then global atomics)
__global__ void bn_stats_kernel(const float* __restrict__ h, float* __restrict__ sums,
                                float* __restrict__ sumsq, int N) {
  __shared__ float ls[HID], ls2[HID];
  int f = threadIdx.x & 63;
  int tg = blockIdx.x * blockDim.x + threadIdx.x;
  int rstart = tg >> 6;
  int rstride = (gridDim.x * blockDim.x) >> 6;
  float s = 0.f, s2 = 0.f;
  for (int r = rstart; r < N; r += rstride) {
    __builtin_prefetch(&h[(r + rstride) * HID + f], 0, 0);  // global_prefetch_b8
    float v = h[r * HID + f];
    s += v; s2 += v * v;
  }
  if (threadIdx.x < HID) { ls[threadIdx.x] = 0.f; ls2[threadIdx.x] = 0.f; }
  __syncthreads();
  atomicAdd(&ls[f], s);
  atomicAdd(&ls2[f], s2);
  __syncthreads();
  if (threadIdx.x < HID) {
    atomicAdd(&sums[threadIdx.x], ls[threadIdx.x]);
    atomicAdd(&sumsq[threadIdx.x], ls2[threadIdx.x]);
  }
}

__global__ void bn_finalize_kernel(const float* __restrict__ sums, const float* __restrict__ sumsq,
                                   const float* __restrict__ g, const float* __restrict__ be,
                                   float* __restrict__ scale, float* __restrict__ shift, int N) {
  int f = threadIdx.x;
  if (f >= HID) return;
  float invN = 1.0f / (float)N;
  float mu  = sums[f] * invN;
  float var = sumsq[f] * invN - mu * mu;
  float sc  = g[f] * rsqrtf(var + 1e-5f);
  scale[f] = sc;
  shift[f] = be[f] - mu * sc;
}

__global__ void bn_relu_kernel(const float* __restrict__ h, const float* __restrict__ scale,
                               const float* __restrict__ shift, float* __restrict__ y, int N) {
  int idx = blockIdx.x * blockDim.x + threadIdx.x;
  if (idx >= N * HID) return;
  int f = idx & 63;
  y[idx] = fmaxf(0.f, h[idx] * scale[f] + shift[f]);
}

__global__ void pool_kernel(const float* __restrict__ x, const int* __restrict__ batch,
                            float* __restrict__ pooled, int N) {
  int idx = blockIdx.x * blockDim.x + threadIdx.x;
  if (idx >= N * HID) return;
  int i = idx >> 6, f = idx & 63;
  atomicAdd(&pooled[batch[i] * HID + f], x[idx]);
}

__global__ void linear_kernel(const float* __restrict__ pooled, const float* __restrict__ W,
                              const float* __restrict__ bias, float* __restrict__ out,
                              int NG, int NCLS) {
  int idx = blockIdx.x * blockDim.x + threadIdx.x;
  if (idx >= NG * NCLS) return;
  int g = idx / NCLS, c = idx % NCLS;
  float s = bias[c];
#pragma unroll
  for (int f = 0; f < HID; ++f) s += pooled[g * HID + f] * W[f * NCLS + c];
  out[idx] = s;
}

extern "C" void kernel_launch(void* const* d_in, const int* in_sizes, int n_in,
                              void* d_out, int out_size, void* d_ws, size_t ws_size,
                              hipStream_t stream) {
  const int*   sid   = (const int*)d_in[0];
  const int*   cid   = (const int*)d_in[1];
  const int*   edges = (const int*)d_in[2];
  const int*   batch = (const int*)d_in[3];
  const float* sW    = (const float*)d_in[4];
  const float* cW    = (const float*)d_in[5];
  const float* W1l   = (const float*)d_in[6];
  const float* W1r   = (const float*)d_in[8];
  const float* W2l   = (const float*)d_in[11];
  const float* W2r   = (const float*)d_in[13];
  const float* b_[2]  = {(const float*)d_in[7],  (const float*)d_in[12]};
  const float* g_[2]  = {(const float*)d_in[9],  (const float*)d_in[14]};
  const float* be_[2] = {(const float*)d_in[10], (const float*)d_in[15]};
  const float* Wlin  = (const float*)d_in[16];
  const float* blin  = (const float*)d_in[17];

  const int N    = in_sizes[0];
  const int E    = in_sizes[2] / 2;
  const int NCLS = in_sizes[17];
  const int NG   = out_size / NCLS;

  float* ws = (float*)d_ws;
  size_t off = 0;
  float* xbuf  = ws + off; off += (size_t)N * HID;
  float* msg   = ws + off; off += (size_t)N * HID;
  float* hbuf  = ws + off; off += (size_t)N * HID;
  float* cnt   = ws + off; off += (size_t)N;
  float* sums  = ws + off; off += HID;       // sums + sumsq contiguous
  float* sumsq = ws + off; off += HID;
  float* scale = ws + off; off += HID;
  float* shift = ws + off; off += HID;
  float* pooled= ws + off; off += (size_t)NG * HID;
  __bf16* pbw  = (__bf16*)(ws + off);        // 4 * 4096 bf16 = 32 KB, 32B-aligned
  (void)ws_size; (void)n_in;

  const int TB = 256;                        // 8 waves / block
  int nTiles = (N + 15) / 16;
  int sageBlocks = (nTiles + (TB / 32) - 1) / (TB / 32);

  // one-time: embed, degree counts, bf16 weight packing
  embed_kernel<<<(N * HID + TB - 1) / TB, TB, 0, stream>>>(sid, cid, sW, cW, xbuf, N);
  hipMemsetAsync(cnt, 0, (size_t)N * sizeof(float), stream);
  count_kernel<<<(E + TB - 1) / TB, TB, 0, stream>>>(edges, cnt, E);
  pack_w_kernel<<<(4 * 4096 + TB - 1) / TB, TB, 0, stream>>>(W1l, W1r, W2l, W2r, pbw);

  for (int l = 0; l < 2; ++l) {
    hipMemsetAsync(msg, 0, (size_t)N * HID * sizeof(float), stream);
    scatter_kernel<<<(E * 32 + TB - 1) / TB, TB, 0, stream>>>(edges, xbuf, msg, E);
    sage_wmma_kernel<<<sageBlocks, TB, 0, stream>>>(msg, cnt, xbuf,
                                                    pbw + (size_t)l * 8192, b_[l], hbuf, N);
    hipMemsetAsync(sums, 0, 2 * HID * sizeof(float), stream);
    bn_stats_kernel<<<1024, TB, 0, stream>>>(hbuf, sums, sumsq, N);
    bn_finalize_kernel<<<1, 64, 0, stream>>>(sums, sumsq, g_[l], be_[l], scale, shift, N);
    bn_relu_kernel<<<(N * HID + TB - 1) / TB, TB, 0, stream>>>(hbuf, scale, shift, xbuf, N);
  }

  hipMemsetAsync(pooled, 0, (size_t)NG * HID * sizeof(float), stream);
  pool_kernel<<<(N * HID + TB - 1) / TB, TB, 0, stream>>>(xbuf, batch, pooled, N);
  linear_kernel<<<(NG * NCLS + TB - 1) / TB, TB, 0, stream>>>(pooled, Wlin, blin,
                                                              (float*)d_out, NG, NCLS);
}
